// ScalarQuantizedXWLinear_82085414961560
// MI455X (gfx1250) — compile-verified, matrix operand order checked by reading, not run on hardware
//
#include <hip/hip_runtime.h>
#include <hip/hip_bf16.h>

typedef __attribute__((ext_vector_type(16))) _Float16 v16h;
typedef __attribute__((ext_vector_type(8)))  _Float16 v8h;
typedef __attribute__((ext_vector_type(4)))  _Float16 v4h;
typedef __attribute__((ext_vector_type(8)))  float    v8f;
typedef __attribute__((ext_vector_type(4)))  int      v4i;

#define QMINF (-128.0f)
#define QMAXF (127.0f)

// ---------------------------------------------------------------------------
// CDNA5 async global->LDS copy (ASYNCcnt-tracked), with safe fallback.
// ---------------------------------------------------------------------------
typedef __attribute__((address_space(1))) v4i gas_v4i;
typedef __attribute__((address_space(3))) v4i las_v4i;

#if __has_builtin(__builtin_amdgcn_global_load_async_to_lds_b128)
#define HAVE_ASYNC_LDS 1
#else
#define HAVE_ASYNC_LDS 0
#endif

__device__ __forceinline__ void copy16_g2l(const void* g, void* l) {
#if HAVE_ASYNC_LDS
  __builtin_amdgcn_global_load_async_to_lds_b128(
      (gas_v4i*)g, (las_v4i*)l, 0, 0);
#else
  *(v8h*)l = *(const v8h*)g;
#endif
}

#if __has_builtin(__builtin_amdgcn_s_wait_asynccnt)
#define ASYNC_WAIT(n) __builtin_amdgcn_s_wait_asynccnt(n)
#else
#define ASYNC_WAIT(n) asm volatile("s_wait_asynccnt %0" ::"i"(n) : "memory")
#endif

// ---------------------------------------------------------------------------
// Kernel 1: quantize hidden_states onto int8 grid, convert to f16.
// ---------------------------------------------------------------------------
__global__ __launch_bounds__(256) void quant_f32_to_f16(
    const float* __restrict__ x, _Float16* __restrict__ xq,
    const float* __restrict__ scale_p, long n4) {
  long i = (long)blockIdx.x * blockDim.x + threadIdx.x;
  if (i >= n4) return;
  float s = fmaxf(scale_p[0], 1e-8f);
  float inv = 1.0f / s;
  float4 v = ((const float4*)x)[i];
  v4h o;
  o[0] = (_Float16)(fminf(fmaxf(rintf(v.x * inv), QMINF), QMAXF) * s);
  o[1] = (_Float16)(fminf(fmaxf(rintf(v.y * inv), QMINF), QMAXF) * s);
  o[2] = (_Float16)(fminf(fmaxf(rintf(v.z * inv), QMINF), QMAXF) * s);
  o[3] = (_Float16)(fminf(fmaxf(rintf(v.w * inv), QMINF), QMAXF) * s);
  ((v4h*)xq)[i] = o;
}

// ---------------------------------------------------------------------------
// Kernel 2: weight (K x N f32) -> Wt (N x K f16) via LDS transpose.
// ---------------------------------------------------------------------------
__global__ __launch_bounds__(256) void wconv_transpose(
    const float* __restrict__ w, _Float16* __restrict__ wt, int K, int N) {
  __shared__ float tile[32][33];
  int tx = threadIdx.x;
  int ty = threadIdx.y;
  int n0 = blockIdx.x * 32;
  int k0 = blockIdx.y * 32;
#pragma unroll
  for (int j = 0; j < 32; j += 8)
    tile[ty + j][tx] = w[(long)(k0 + ty + j) * N + (n0 + tx)];
  __syncthreads();
#pragma unroll
  for (int j = 0; j < 32; j += 8)
    wt[(long)(n0 + ty + j) * K + (k0 + tx)] = (_Float16)tile[tx][ty + j];
}

// ---------------------------------------------------------------------------
// Kernel 3: f16 WMMA GEMM, f32 accumulation, bias epilogue.
// Block tile 256(M) x 128(N), BK=32, 8 wave32 waves in 4(M)x2(N), 64x64/wave.
// 3-stage LDS pipeline fed by async global->LDS b128 (one barrier per step).
// ---------------------------------------------------------------------------
#define BLK_M 256
#define BLK_N 128
#define BK 32
#define LDS_STRIDE 40   // BK + 8 f16 pad (80B rows) to spread LDS banks

union FragH { v16h v; v8h h[2]; };

// Per-wave async ops per stage: 4 (A) + 2 (B) = 6.
__device__ __forceinline__ void issue_tile_copies(
    const _Float16* __restrict__ A, const _Float16* __restrict__ Bt,
    _Float16* sa, _Float16* sb, long bm0, long bn0, int k0, int K, int tid) {
#pragma unroll
  for (int c = 0; c < 4; ++c) {          // A tile: 256 rows x 4 b128 chunks
    int idx = tid + c * 256;
    int row = idx >> 2;
    int cc  = idx & 3;
    copy16_g2l(A + (bm0 + row) * (long)K + k0 + cc * 8,
               sa + row * LDS_STRIDE + cc * 8);
  }
#pragma unroll
  for (int c = 0; c < 2; ++c) {          // B tile: 128 rows x 4 b128 chunks
    int idx = tid + c * 256;
    int row = idx >> 2;
    int cc  = idx & 3;
    copy16_g2l(Bt + (bn0 + row) * (long)K + k0 + cc * 8,
               sb + row * LDS_STRIDE + cc * 8);
  }
}

__global__ __launch_bounds__(256) void gemm_f16_wmma(
    const _Float16* __restrict__ A,   // M x K f16 (quantized activations)
    const _Float16* __restrict__ Bt,  // N x K f16 (transposed weights)
    const float* __restrict__ bias,   // N
    float* __restrict__ C,            // M x N f32
    int M, int N, int K) {
  __shared__ _Float16 sA[3][BLK_M * LDS_STRIDE];   // 3 x 20 KB
  __shared__ _Float16 sB[3][BLK_N * LDS_STRIDE];   // 3 x 10 KB

  const int tid   = threadIdx.x;
  const int wave  = tid >> 5;
  const int lane  = tid & 31;
  const int wm    = wave & 3;    // M slot: 0..3 (x64 rows)
  const int wn    = wave >> 2;   // N slot: 0..1 (x64 cols)
  const int l16   = lane & 15;
  const int lhalf = lane >> 4;   // 0: K 0-7/16-23 ; 1: K 8-15/24-31
  const int koff  = lhalf * 8;

  const long bm0 = (long)blockIdx.y * BLK_M;
  const long bn0 = (long)blockIdx.x * BLK_N;

  v8f acc[4][4];
#pragma unroll
  for (int i = 0; i < 4; ++i)
#pragma unroll
    for (int j = 0; j < 4; ++j)
      acc[i][j] = (v8f){0.f, 0.f, 0.f, 0.f, 0.f, 0.f, 0.f, 0.f};

  // Prologue: prefetch stages 0 and 1.
  issue_tile_copies(A, Bt, sA[0], sB[0], bm0, bn0, 0, K, tid);
  if (BK < K)
    issue_tile_copies(A, Bt, sA[1], sB[1], bm0, bn0, BK, K, tid);

  int cur = 0;
  for (int k0 = 0; k0 < K; k0 += BK) {
    // 1) my stage-i copies landed (<=1 later stage may remain in flight)
    if (k0 + BK < K) ASYNC_WAIT(6);
    else             ASYNC_WAIT(0);
    // 2) everyone's stage-i copies landed AND everyone finished stage i-1
    __syncthreads();
    // 3) refill the buffer consumed at stage i-1 with stage i+2
    if (k0 + 2 * BK < K) {
      int b2 = cur + 2;
      if (b2 >= 3) b2 -= 3;
      issue_tile_copies(A, Bt, sA[b2], sB[b2], bm0, bn0, k0 + 2 * BK, K, tid);
    }

    // 4) compute stage i: preload all 8 fragments, then 16 WMMAs
    FragH bf[4];
#pragma unroll
    for (int tn = 0; tn < 4; ++tn) {
      const _Float16* bp = &sB[cur][(wn * 64 + tn * 16 + l16) * LDS_STRIDE + koff];
      bf[tn].h[0] = *(const v8h*)(bp);        // K = koff .. koff+7
      bf[tn].h[1] = *(const v8h*)(bp + 16);   // K = koff+16 .. koff+23
    }
    FragH af[4];
#pragma unroll
    for (int tm = 0; tm < 4; ++tm) {
      const _Float16* ap = &sA[cur][(wm * 64 + tm * 16 + l16) * LDS_STRIDE + koff];
      af[tm].h[0] = *(const v8h*)(ap);
      af[tm].h[1] = *(const v8h*)(ap + 16);
    }
#pragma unroll
    for (int tm = 0; tm < 4; ++tm)
#pragma unroll
      for (int tn = 0; tn < 4; ++tn)
        acc[tm][tn] = __builtin_amdgcn_wmma_f32_16x16x32_f16(
            false, af[tm].v, false, bf[tn].v, (short)0, acc[tm][tn], false, false);

    if (++cur >= 3) cur = 0;
  }

  // ---- epilogue: bias + store (C/D layout: VGPR r -> M=r / M=r+8) ----
#pragma unroll
  for (int tm = 0; tm < 4; ++tm) {
#pragma unroll
    for (int tn = 0; tn < 4; ++tn) {
      long n = bn0 + wn * 64 + tn * 16 + l16;
      float bv = bias[n];
      long mbase = bm0 + wm * 64 + tm * 16 + lhalf * 8;
      v8f v = acc[tm][tn];
#pragma unroll
      for (int r = 0; r < 8; ++r)
        C[(mbase + r) * N + n] = v[r] + bv;
    }
  }
}

// ---------------------------------------------------------------------------
// Launch
// ---------------------------------------------------------------------------
extern "C" void kernel_launch(void* const* d_in, const int* in_sizes, int n_in,
                              void* d_out, int out_size, void* d_ws, size_t ws_size,
                              hipStream_t stream) {
  const float* hidden = (const float*)d_in[0];   // (B,S,D) f32
  const float* weight = (const float*)d_in[1];   // (D,DOUT) f32
  const float* xscale = (const float*)d_in[2];   // scalar f32
  const float* bias   = (const float*)d_in[3];   // (DOUT) f32
  float* out = (float*)d_out;

  const int N = in_sizes[3];                 // DOUT
  const int K = in_sizes[1] / N;             // D
  const int M = in_sizes[0] / K;             // B*S

  _Float16* Aq = (_Float16*)d_ws;                              // M*K f16
  _Float16* Wt = (_Float16*)((char*)d_ws + (size_t)M * K * 2); // N*K f16

  long n4 = (long)M * K / 4;
  quant_f32_to_f16<<<dim3((unsigned)((n4 + 255) / 256)), dim3(256), 0, stream>>>(
      hidden, Aq, xscale, n4);

  wconv_transpose<<<dim3(N / 32, K / 32), dim3(32, 8), 0, stream>>>(
      weight, Wt, K, N);

  gemm_f16_wmma<<<dim3(N / BLK_N, M / BLK_M), dim3(256), 0, stream>>>(
      Aq, Wt, bias, out, M, N, K);
}
